// GCN_5299989643753
// MI455X (gfx1250) — matrix operand order repaired and checked by reading, not compiled, not run on hardware
//
#include <hip/hip_runtime.h>
#include <hip/hip_bf16.h>

#define DF 128
#define NGRAPH 512
#define NCLS 10
#define MTILE 64   // rows per block in the WMMA GEMM

typedef __attribute__((ext_vector_type(16))) __bf16 v16bf;
typedef __attribute__((ext_vector_type(8)))  __bf16 v8bf;
typedef __attribute__((ext_vector_type(4)))  __bf16 v4bf;
typedef __attribute__((ext_vector_type(8)))  float  v8f;

// ---------------- elementwise helpers ----------------
__global__ void fill_kernel(float* __restrict__ p, float v, int n) {
    int i = blockIdx.x * blockDim.x + threadIdx.x;
    if (i < n) p[i] = v;
}

__global__ void deg_kernel(const int* __restrict__ dst, float* __restrict__ deg, int n_edges) {
    int e = blockIdx.x * blockDim.x + threadIdx.x;
    if (e < n_edges) atomicAdd(&deg[dst[e]], 1.0f);
}

__global__ void rsqrt_kernel(const float* __restrict__ deg, float* __restrict__ dinv, int n) {
    int i = blockIdx.x * blockDim.x + threadIdx.x;
    if (i < n) dinv[i] = rsqrtf(deg[i]);
}

// agg = h * dinv[node]^2   (self-loop term), float4-vectorized
__global__ void selfloop_kernel(const float4* __restrict__ h, float4* __restrict__ agg,
                                const float* __restrict__ dinv, int n_nodes) {
    int idx = blockIdx.x * blockDim.x + threadIdx.x;     // over n_nodes * 32 float4s
    if (idx >= n_nodes * 32) return;
    int node = idx >> 5;
    float s = dinv[node]; s = s * s;
    float4 v = h[idx];
    v.x *= s; v.y *= s; v.z *= s; v.w *= s;
    agg[idx] = v;
}

// agg[dst] += h[src] * dinv[src]*dinv[dst]  — one wave32 per edge, 4 floats/lane.
// h+agg (~102 MB) fit in the 192 MB L2, so these f32 atomics stay L2-resident.
__global__ __launch_bounds__(256) void scatter_kernel(const float4* __restrict__ h,
                                                      float* __restrict__ agg,
                                                      const int* __restrict__ src,
                                                      const int* __restrict__ dst,
                                                      const float* __restrict__ dinv,
                                                      int n_edges) {
    int e = blockIdx.x * 8 + (threadIdx.x >> 5);
    if (e >= n_edges) return;
    int lane = threadIdx.x & 31;
    int s = src[e], d = dst[e];
    float w = dinv[s] * dinv[d];
    float4 v = h[(size_t)s * 32 + lane];
    float* a = agg + (size_t)d * DF + lane * 4;
    atomicAdd(a + 0, v.x * w);
    atomicAdd(a + 1, v.y * w);
    atomicAdd(a + 2, v.z * w);
    atomicAdd(a + 3, v.w * w);
}

__global__ void bias_relu_kernel(const float* __restrict__ agg, const float* __restrict__ bias,
                                 float* __restrict__ out, int n_elems) {
    int idx = blockIdx.x * blockDim.x + threadIdx.x;
    if (idx >= n_elems) return;
    float v = agg[idx] + bias[idx & (DF - 1)];
    out[idx] = v > 0.f ? v : 0.f;
}

// ---------------- WMMA GEMM: out[n_rows x 128] = in[n_rows x 128] @ W[128 x 128] ----------------
// Block = 256 thr = 8 waves; block owns a 64-row M tile (4 sub-tiles of 16).
// Wave w owns N columns [16w,16w+16). bf16 conversion happens ONCE at staging;
// A fragments are ds_load_b128 pairs, B is gathered per element. 16 x v_wmma_f32_16x16x32_bf16 per wave,
// with each B fragment reused across the 4 M sub-tiles.
__global__ __launch_bounds__(256) void gemm_bf16_wmma(const float* __restrict__ in,
                                                      const float* __restrict__ W,
                                                      float* __restrict__ out, int n_rows) {
    __shared__ __align__(16) __bf16 xs[MTILE][DF + 8];   // bf16 x tile  (~17 KB)
    __shared__ __align__(16) __bf16 wsh[DF][DF + 8];     // bf16 W       (~34 KB)

    const int tid = threadIdx.x;
    const int m_base = blockIdx.x * MTILE;

    // stage x tile: MTILE x 128 fp32 -> bf16, float4 in / v4bf out (ds_store_b64)
    for (int t = tid; t < MTILE * 32; t += 256) {
        int r = t >> 5, c4 = t & 31;
        float4 v = make_float4(0.f, 0.f, 0.f, 0.f);
        int m = m_base + r;
        if (m < n_rows) v = ((const float4*)in)[(size_t)m * 32 + c4];
        v4bf p;
        p[0] = (__bf16)v.x; p[1] = (__bf16)v.y; p[2] = (__bf16)v.z; p[3] = (__bf16)v.w;
        *(v4bf*)&xs[r][c4 * 4] = p;
    }
    // stage W: 128x128 fp32 -> bf16
    for (int t = tid; t < DF * 32; t += 256) {
        int r = t >> 5, c4 = t & 31;
        float4 v = ((const float4*)W)[(size_t)r * 32 + c4];
        v4bf p;
        p[0] = (__bf16)v.x; p[1] = (__bf16)v.y; p[2] = (__bf16)v.z; p[3] = (__bf16)v.w;
        *(v4bf*)&wsh[r][c4 * 4] = p;
    }
    __syncthreads();

    const int wave  = tid >> 5;      // N tile index 0..7
    const int lane  = tid & 31;
    const int lhalf = lane >> 4;     // 0/1
    const int lmod  = lane & 15;

    v8f c[4] = {{}, {}, {}, {}};
#pragma unroll
    for (int kt = 0; kt < 4; ++kt) {
        // B 32x16 bf16: lane holds N=lmod; elems 0..7 -> K=base..base+7,
        // elems 8..15 -> K=base+16..base+23, base = kt*32 + lhalf*8
        v16bf b;
#pragma unroll
        for (int s = 0; s < 16; ++s)
            b[s] = wsh[kt * 32 + (s < 8 ? lhalf * 8 + s : 16 + lhalf * 8 + (s - 8))][wave * 16 + lmod];
#pragma unroll
        for (int mt = 0; mt < 4; ++mt) {
            // A 16x32 bf16: lane holds M=lmod; elems 0..7 -> K=base..base+7,
            // elems 8..15 -> K=base+16..base+23, base = kt*32 + lhalf*8
            const v8bf* rowA = (const v8bf*)&xs[mt * 16 + lmod][0];
            int kc = kt * 4 + lhalf;          // 8-elem chunk index
            v8bf alo = rowA[kc];
            v8bf ahi = rowA[kc + 2];
            v16bf a = __builtin_shufflevector(alo, ahi,
                                              0, 1, 2, 3, 4, 5, 6, 7,
                                              8, 9, 10, 11, 12, 13, 14, 15);
            c[mt] = __builtin_amdgcn_wmma_f32_16x16x32_bf16(false, a, false, b,
                                                            (short)0, c[mt], false, false);
        }
    }
    // C/D f32 16x16: VGPR r -> M = r + 8*lhalf, N = lmod
#pragma unroll
    for (int mt = 0; mt < 4; ++mt) {
#pragma unroll
        for (int r = 0; r < 8; ++r) {
            int m = m_base + mt * 16 + r + lhalf * 8;
            if (m < n_rows) out[(size_t)m * DF + wave * 16 + lmod] = c[mt][r];
        }
    }
}

// ---------------- pooling + classifier ----------------
__global__ __launch_bounds__(256) void pool_kernel(const float4* __restrict__ h,
                                                   const int* __restrict__ batch,
                                                   float* __restrict__ pooled,
                                                   float* __restrict__ cnt, int n_nodes) {
    int node = blockIdx.x * 8 + (threadIdx.x >> 5);
    if (node >= n_nodes) return;
    int lane = threadIdx.x & 31;
    int g = batch[node];
    float4 v = h[(size_t)node * 32 + lane];
    float* p = pooled + (size_t)g * DF + lane * 4;
    atomicAdd(p + 0, v.x);
    atomicAdd(p + 1, v.y);
    atomicAdd(p + 2, v.z);
    atomicAdd(p + 3, v.w);
    if (lane == 0) atomicAdd(&cnt[g], 1.0f);
}

__global__ void cls_kernel(const float* __restrict__ pooled, const float* __restrict__ cnt,
                           const float* __restrict__ Wc, const float* __restrict__ bc,
                           float* __restrict__ out) {
    int t = blockIdx.x * blockDim.x + threadIdx.x;
    if (t >= NGRAPH * NCLS) return;
    int g = t / NCLS, c = t % NCLS;
    float inv = 1.0f / fmaxf(cnt[g], 1.0f);
    float acc = bc[c];
#pragma unroll 8
    for (int f = 0; f < DF; ++f)
        acc += pooled[g * DF + f] * inv * Wc[f * NCLS + c];
    out[t] = acc;
}

// ---------------- host orchestration ----------------
extern "C" void kernel_launch(void* const* d_in, const int* in_sizes, int n_in,
                              void* d_out, int out_size, void* d_ws, size_t ws_size,
                              hipStream_t stream) {
    const float* x     = (const float*)d_in[0];
    const int*   esrc  = (const int*)d_in[1];
    const int*   edst  = (const int*)d_in[2];
    const int*   batch = (const int*)d_in[3];
    const float* Wl[3] = {(const float*)d_in[4], (const float*)d_in[6], (const float*)d_in[8]};
    const float* bl[3] = {(const float*)d_in[5], (const float*)d_in[7], (const float*)d_in[9]};
    const float* Wc    = (const float*)d_in[10];
    const float* bc    = (const float*)d_in[11];

    const int n_nodes = in_sizes[0] / DF;   // 100000
    const int n_edges = in_sizes[1];        // 1600000

    // workspace layout (floats); all region starts stay 16B aligned
    float* ws   = (float*)d_ws;
    size_t off  = 0;
    float* deg    = ws + off; off += (size_t)n_nodes;
    float* dinv   = ws + off; off += (size_t)n_nodes;
    float* h      = ws + off; off += (size_t)n_nodes * DF;
    float* agg    = ws + off; off += (size_t)n_nodes * DF;
    float* pooled = ws + off; off += (size_t)NGRAPH * DF;
    float* cnt    = ws + off; off += (size_t)NGRAPH;

    // degree (with self-loop) and symmetric normalization
    fill_kernel<<<(n_nodes + 255) / 256, 256, 0, stream>>>(deg, 1.0f, n_nodes);
    deg_kernel<<<(n_edges + 255) / 256, 256, 0, stream>>>(edst, deg, n_edges);
    rsqrt_kernel<<<(n_nodes + 255) / 256, 256, 0, stream>>>(deg, dinv, n_nodes);

    const int gemm_blocks = (n_nodes + MTILE - 1) / MTILE;
    const int elem_n      = n_nodes * DF;

    const float* cur = x;
    for (int l = 0; l < 3; ++l) {
        gemm_bf16_wmma<<<gemm_blocks, 256, 0, stream>>>(cur, Wl[l], h, n_nodes);
        selfloop_kernel<<<(n_nodes * 32 + 255) / 256, 256, 0, stream>>>(
            (const float4*)h, (float4*)agg, dinv, n_nodes);
        scatter_kernel<<<(n_edges + 7) / 8, 256, 0, stream>>>(
            (const float4*)h, agg, esrc, edst, dinv, n_edges);
        bias_relu_kernel<<<(elem_n + 255) / 256, 256, 0, stream>>>(agg, bl[l], h, elem_n);
        cur = h;  // in-place safe: gemm stages its 64 rows in LDS before writing them
    }

    fill_kernel<<<(NGRAPH * DF + 255) / 256, 256, 0, stream>>>(pooled, 0.f, NGRAPH * DF);
    fill_kernel<<<(NGRAPH + 255) / 256, 256, 0, stream>>>(cnt, 0.f, NGRAPH);
    pool_kernel<<<(n_nodes + 7) / 8, 256, 0, stream>>>(
        (const float4*)h, batch, pooled, cnt, n_nodes);
    cls_kernel<<<(NGRAPH * NCLS + 255) / 256, 256, 0, stream>>>(
        pooled, cnt, Wc, bc, (float*)d_out);
}
